// GRUmodel_87608742904505
// MI455X (gfx1250) — matrix operand order repaired and checked by reading, not compiled, not run on hardware
//
#include <hip/hip_runtime.h>

// ---------------------------------------------------------------------------
// GRU on MI455X (gfx1250): bf16 WMMA (v_wmma_f32_16x16x32_bf16) GEMMs with
// fp32 accumulate. The sequential scan is fully fused: one kernel per step
// computes r/z/n gate tiles in registers (4 WMMA accumulators per wave) and
// applies the GRU update in-register — no gate scratch traffic, one launch
// per timestep. bf16 hidden state is ping-ponged to avoid read/write races.
// ---------------------------------------------------------------------------

typedef __bf16 bf16;
typedef __attribute__((ext_vector_type(16))) __bf16 v16bf;
typedef __attribute__((ext_vector_type(8)))  __bf16 v8bf;
typedef __attribute__((ext_vector_type(8)))  float  v8f;

#define DI static __device__ __forceinline__

constexpr int B = 64, T = 512, I = 512, H = 1024, O = 512;
constexpr int G = 3 * H;

constexpr size_t NX    = (size_t)B * T * I; // 16,777,216
constexpr size_t NWIH  = (size_t)G * I;     //  1,572,864
constexpr size_t NWHH  = (size_t)G * H;     //  3,145,728
constexpr size_t NWOUT = (size_t)O * H;     //    524,288
constexpr size_t NH    = (size_t)B * H;     //     65,536
constexpr size_t NHS   = (size_t)B * T * H; // 33,554,432

// ---------------------------------------------------------------------------
// WMMA fragment helpers (16-bit A 16x32 / B 32x16; contiguous k = 16*(lane/16) + s, same in A and B)
// ---------------------------------------------------------------------------

union ABu { v16bf v; v8bf h[2]; };

// A fragment: row m = m0 + (lane&15); lanes 0-15 hold K 0..15,
// lanes 16-31 hold K 16..31  -> one contiguous 32B load per lane.
DI v16bf load_a_frag(const bf16* __restrict__ row, int k0, int lane) {
  const int sel = (lane >> 4) & 1;
  return *(const v16bf*)(row + k0 + sel * 16);
}

// B fragment from row-major W[N][K]: col n = n0 + (lane&15); lanes 0-15 hold
// K=k0..k0+15, lanes 16-31 hold K=k0+16..k0+31 -> one contiguous 32B load.
DI v16bf load_b_frag(const bf16* __restrict__ row, int k0, int lane) {
  const int sel = (lane >> 4) & 1;
  return *(const v16bf*)(row + k0 + sel * 16);
}

DI v8f wmma_bf16(v16bf a, v16bf b, v8f c) {
  // (neg_a, A, neg_b, B, c_mod, C, reuse_a, reuse_b)
  return __builtin_amdgcn_wmma_f32_16x16x32_bf16(false, a, false, b,
                                                 (short)0, c, false, false);
}

DI float fsig(float x)  { return 1.0f / (1.0f + __expf(-x)); }
DI float ftanh(float x) {
  x = fminf(15.0f, fmaxf(-15.0f, x));
  float e = __expf(2.0f * x);
  return (e - 1.0f) / (e + 1.0f);
}

// ---------------------------------------------------------------------------
// Kernels
// ---------------------------------------------------------------------------

// One-time fp32 -> bf16 conversion of X and weights; h0 init (fp32 + bf16).
__global__ __launch_bounds__(256) void prep_kernel(
    const float* __restrict__ X,   const float* __restrict__ wih,
    const float* __restrict__ whh, const float* __restrict__ wout,
    const float* __restrict__ h0,
    bf16* __restrict__ Xb,   bf16* __restrict__ Wihb,
    bf16* __restrict__ Whhb, bf16* __restrict__ Woutb,
    float* __restrict__ hf,  bf16* __restrict__ hb0) {
  const size_t tid    = (size_t)blockIdx.x * blockDim.x + threadIdx.x;
  const size_t stride = (size_t)gridDim.x * blockDim.x;
  for (size_t i = tid; i < NX;    i += stride) Xb[i]    = (bf16)X[i];
  for (size_t i = tid; i < NWIH;  i += stride) Wihb[i]  = (bf16)wih[i];
  for (size_t i = tid; i < NWHH;  i += stride) Whhb[i]  = (bf16)whh[i];
  for (size_t i = tid; i < NWOUT; i += stride) Woutb[i] = (bf16)wout[i];
  for (size_t i = tid; i < NH;    i += stride) {
    float v = h0[i];
    hf[i]  = v;
    hb0[i] = (bf16)v;
  }
}

// Fully fused GRU timestep. Wave tile: 16 batch rows x 16 hidden cols.
// 4 accumulators: fused r (x+h), fused z (x+h), xn, hn (kept separate for
// PyTorch semantics n = tanh(xn + r*hn) with b_hh inside hn).
// 256 waves -> 32 blocks x 8 waves; Wih/Whh (9 MB bf16) are L2-resident.
__global__ __launch_bounds__(256) void gru_step_fused(
    const bf16* __restrict__ Xb,    // [B, T, I]
    const bf16* __restrict__ hbin,  // [B, H] bf16 prev state (read-only)
    const bf16* __restrict__ Wih,   // [G, I] row-major
    const bf16* __restrict__ Whh,   // [G, H] row-major
    const float* __restrict__ bih,  const float* __restrict__ bhh,
    float* __restrict__ hf,         // [B, H] fp32 state (per-lane disjoint)
    bf16* __restrict__ hbout,       // [B, H] bf16 next state
    bf16* __restrict__ hs,          // [B*T, H] history for output GEMM
    int t) {
  const int lane = threadIdx.x & 31;
  const int gw   = blockIdx.x * 8 + (threadIdx.x >> 5); // 0..255
  const int m0   = (gw & 3) * 16;                       // batch tile
  const int n0   = (gw >> 2) * 16;                      // hidden-column tile

  v8f acc_r = {}, acc_z = {}, acc_xn = {}, acc_hn = {};

  { // x-part: A = X[:, t, :] (K = I); B rows = Wih[{0,H,2H} + n0 + c]
    const bf16* arow = Xb + ((size_t)(m0 + (lane & 15)) * T + t) * I;
    const bf16* br = Wih + (size_t)(n0 + (lane & 15)) * I;
    const bf16* bz = br + (size_t)H * I;
    const bf16* bn = bz + (size_t)H * I;
#pragma unroll 2
    for (int k = 0; k < I; k += 32) {
      v16bf a = load_a_frag(arow, k, lane);
      acc_r  = wmma_bf16(a, load_b_frag(br, k, lane), acc_r);
      acc_z  = wmma_bf16(a, load_b_frag(bz, k, lane), acc_z);
      acc_xn = wmma_bf16(a, load_b_frag(bn, k, lane), acc_xn);
    }
  }
  { // h-part: A = h_prev (K = H); B rows = Whh[{0,H,2H} + n0 + c]
    const bf16* arow = hbin + (size_t)(m0 + (lane & 15)) * H;
    const bf16* br = Whh + (size_t)(n0 + (lane & 15)) * H;
    const bf16* bz = br + (size_t)H * H;
    const bf16* bn = bz + (size_t)H * H;
#pragma unroll 2
    for (int k = 0; k < H; k += 32) {
      v16bf a = load_a_frag(arow, k, lane);
      acc_r  = wmma_bf16(a, load_b_frag(br, k, lane), acc_r);
      acc_z  = wmma_bf16(a, load_b_frag(bz, k, lane), acc_z);
      acc_hn = wmma_bf16(a, load_b_frag(bn, k, lane), acc_hn);
    }
  }

  // In-register GRU update. C/D layout: reg rr -> row rbase+rr, lane -> col.
  const int col   = n0 + (lane & 15);
  const int rbase = m0 + ((lane >> 4) << 3);
  const float br_ = bih[col]         + bhh[col];
  const float bz_ = bih[H + col]     + bhh[H + col];
  const float bxn = bih[2 * H + col];
  const float bhn = bhh[2 * H + col];

#pragma unroll
  for (int rr = 0; rr < 8; ++rr) {
    const int row = rbase + rr;
    const float r = fsig(acc_r[rr] + br_);
    const float z = fsig(acc_z[rr] + bz_);
    const float n = ftanh(acc_xn[rr] + bxn + r * (acc_hn[rr] + bhn));
    const float hprev = hf[(size_t)row * H + col];
    const float hnew  = (1.0f - z) * n + z * hprev;
    hf[(size_t)row * H + col]           = hnew;
    hbout[(size_t)row * H + col]        = (bf16)hnew;
    hs[((size_t)row * T + t) * H + col] = (bf16)hnew;
  }
}

// out[B*T, O] = hs @ Wout^T + b_out. Large-M GEMM: 16x64 per wave (A reuse x4).
__global__ __launch_bounds__(256) void out_gemm(
    const bf16* __restrict__ hs,    // [B*T, H]
    const bf16* __restrict__ Wout,  // [O, H] row-major
    const float* __restrict__ bout, float* __restrict__ out) {
  const int lane = threadIdx.x & 31;
  const int w    = threadIdx.x >> 5;
  const int m0   = blockIdx.x * 128 + w * 16;  // grid.x = (B*T)/128 = 256
  const int n0   = blockIdx.y * 64;            // grid.y = O/64 = 8

  v8f acc[4] = {};
  {
    const bf16* arow  = hs + (size_t)(m0 + (lane & 15)) * H;
    const bf16* brow0 = Wout + (size_t)(n0 + (lane & 15)) * H;
#pragma unroll 2
    for (int k = 0; k < H; k += 32) {
      v16bf a = load_a_frag(arow, k, lane);
#pragma unroll
      for (int j = 0; j < 4; ++j) {
        v16bf b = load_b_frag(brow0 + (size_t)(j * 16) * H, k, lane);
        acc[j] = wmma_bf16(a, b, acc[j]);
      }
    }
  }

  const int col0  = n0 + (lane & 15);
  const int rbase = m0 + ((lane >> 4) << 3);
#pragma unroll
  for (int j = 0; j < 4; ++j) {
    const int col  = col0 + j * 16;
    const float bb = bout[col];
#pragma unroll
    for (int r = 0; r < 8; ++r)
      out[(size_t)(rbase + r) * O + col] = acc[j][r] + bb;
  }
}

__global__ __launch_bounds__(256) void copy_hlast(const float* __restrict__ hf,
                                                  float* __restrict__ dst) {
  const int i = blockIdx.x * blockDim.x + threadIdx.x; // B*H threads
  dst[i] = hf[i];
}

// ---------------------------------------------------------------------------
// Host entry
// ---------------------------------------------------------------------------

extern "C" void kernel_launch(void* const* d_in, const int* in_sizes, int n_in,
                              void* d_out, int out_size, void* d_ws, size_t ws_size,
                              hipStream_t stream) {
  const float* X    = (const float*)d_in[0];
  const float* h0   = (const float*)d_in[1];
  const float* wih  = (const float*)d_in[2];
  const float* whh  = (const float*)d_in[3];
  const float* bih  = (const float*)d_in[4];
  const float* bhh  = (const float*)d_in[5];
  const float* wout = (const float*)d_in[6];
  const float* bout = (const float*)d_in[7];

  float* out   = (float*)d_out;
  float* hlast = out + (size_t)B * T * O;   // tuple: (out[B,T,O], h_last[1,B,H])

  // Workspace carve-out (~111 MB total, 256B-aligned regions).
  char* ws = (char*)d_ws;
  auto carve = [&](size_t bytes) {
    char* p = ws;
    ws += (bytes + 255) & ~(size_t)255;
    return p;
  };
  bf16*  Xb    = (bf16*)carve(NX * sizeof(bf16));
  bf16*  Wihb  = (bf16*)carve(NWIH * sizeof(bf16));
  bf16*  Whhb  = (bf16*)carve(NWHH * sizeof(bf16));
  bf16*  Woutb = (bf16*)carve(NWOUT * sizeof(bf16));
  bf16*  hs    = (bf16*)carve(NHS * sizeof(bf16));
  float* hf    = (float*)carve(NH * sizeof(float));
  bf16*  hb0   = (bf16*)carve(NH * sizeof(bf16));
  bf16*  hb1   = (bf16*)carve(NH * sizeof(bf16));

  prep_kernel<<<2048, 256, 0, stream>>>(X, wih, whh, wout, h0,
                                        Xb, Wihb, Whhb, Woutb, hf, hb0);

  for (int t = 0; t < T; ++t) {
    const bf16* hin  = (t & 1) ? hb1 : hb0;  // ping-pong bf16 state
    bf16*       hout = (t & 1) ? hb0 : hb1;
    gru_step_fused<<<32, 256, 0, stream>>>(Xb, hin, Wihb, Whhb, bih, bhh,
                                           hf, hout, hs, t);
  }

  out_gemm<<<dim3((B * T) / 128, O / 64), 256, 0, stream>>>(hs, Woutb, bout, out);
  copy_hlast<<<(B * H) / 256, 256, 0, stream>>>(hf, hlast);
}
